// WeightedAggregation_42391327212213
// MI455X (gfx1250) — matrix operand order repaired and checked by reading, not compiled, not run on hardware
//
#include <hip/hip_runtime.h>

// Shapes from the reference
#define B_  32
#define N_  32
#define L_  512
#define M_  64
#define F_  64
#define EPS_ 1e-8f

typedef __attribute__((ext_vector_type(16))) _Float16 v16h;
typedef __attribute__((ext_vector_type(8)))  _Float16 v8h;
typedef __attribute__((ext_vector_type(2)))  _Float16 v2h;
typedef __attribute__((ext_vector_type(8)))  float    v8f;

#define LDS_S 40   // halves per f-row of the transposed x chunk (40*2=80B rows: 16B aligned, conflict-friendly)

__device__ __forceinline__ float sigmoidf_fast(float x) {
    return 1.0f / (1.0f + __expf(-x));
}

// ---------------------------------------------------------------------------
// Kernel 1: alphaT[n][m][l] (f16) and recip[n][m] = 1/(sum_l alpha + eps) (f32)
// One block per (n,m) row; 256 threads x 2 l-values each.
// ---------------------------------------------------------------------------
__global__ __launch_bounds__(256)
void wagg_alpha_kernel(const float* __restrict__ t,
                       const float* __restrict__ t_left,
                       const float* __restrict__ t_right,
                       const float* __restrict__ kappa,
                       _Float16* __restrict__ alphaT,
                       float* __restrict__ recip)
{
    __shared__ float red[256];
    const int nm = blockIdx.x;          // n*M + m, 0..2047
    const int n  = nm >> 6;
    const float tl = t_left[nm];
    const float tr = t_right[nm];
    const float kp = kappa[n];
    // stable softplus
    const float k    = fmaxf(kp, 0.0f) + log1pf(__expf(-fabsf(kp)));
    const float invk = 1.0f / k;

    const int tid = threadIdx.x;
    float s = 0.0f;
#pragma unroll
    for (int i = 0; i < 2; ++i) {
        const int l = tid + i * 256;
        const float tv = t[l];
        const float al = sigmoidf_fast((tr - tv) * invk) * sigmoidf_fast((tv - tl) * invk);
        alphaT[(size_t)nm * L_ + l] = (_Float16)al;
        s += al;
    }
    red[tid] = s;
    __syncthreads();
#pragma unroll
    for (int off = 128; off > 0; off >>= 1) {
        if (tid < off) red[tid] += red[tid + off];
        __syncthreads();
    }
    if (tid == 0) recip[nm] = 1.0f / (red[0] + EPS_);
}

// ---------------------------------------------------------------------------
// Kernel 2: out[b,n,m,f] = recip[n,m] * sum_l alphaT[n,m,l] * x[b,n,l,f]
// One block per (b,n): 8 wave32s, each wave = one 16-row M tile x two 16-col
// F tiles of the 64x64 output. K-loop over L in chunks of 32 with
// v_wmma_f32_16x16x32_f16.
// ---------------------------------------------------------------------------
__global__ __launch_bounds__(256)
void wagg_gemm_kernel(const float* __restrict__ x,
                      const _Float16* __restrict__ alphaT,
                      const float* __restrict__ recip,
                      float* __restrict__ out)
{
    __shared__ alignas(16) _Float16 xT[F_ * LDS_S];   // transposed f16 x chunk: [f][k]

    const int bn   = blockIdx.x;        // b*N + n
    const int n    = bn & (N_ - 1);
    const int tid  = threadIdx.x;
    const int lane = tid & 31;
    const int wave = tid >> 5;

    const int m0 = (wave & 3) * 16;     // M tile of this wave
    const int f0 = (wave >> 2) * 32;    // two F tiles: [f0, f0+16) and [f0+16, f0+32)

    const float* xbase = x + (size_t)bn * L_ * F_;

    // A-fragment base: lane<16 holds K {0..15}, lane>=16 holds {16..31}
    const _Float16* abase =
        alphaT + ((size_t)n * M_ + m0 + (lane & 15)) * L_ + ((lane >> 4) * 16);

    // Cooperative staging map: lane group covers one l-row of 64 f contiguously
    const int fgrp = (tid & 15) * 4;    // 16 groups x 4 floats = 64 f
    const int lp   = (tid >> 4) * 2;    // 16 pairs x 2 l = 32 l (even -> packed b32 LDS store)

    // B-fragment pointers (contiguous k order, same as A: lane<16 K0..15, lane>=16 K16..31)
    const int brow = lane & 15;
    const int koff = (lane >> 4) * 16;
    const _Float16* bptr0 = &xT[(f0 + brow) * LDS_S + koff];
    const _Float16* bptr1 = &xT[(f0 + 16 + brow) * LDS_S + koff];

    v8f acc0 = {};
    v8f acc1 = {};

    for (int l0 = 0; l0 < L_; l0 += 32) {
        __syncthreads();   // previous chunk's B reads done
        {
            const float4 r0 = *(const float4*)(xbase + (size_t)(l0 + lp) * F_ + fgrp);
            const float4 r1 = *(const float4*)(xbase + (size_t)(l0 + lp + 1) * F_ + fgrp);
            const float a0[4] = { r0.x, r0.y, r0.z, r0.w };
            const float a1[4] = { r1.x, r1.y, r1.z, r1.w };
#pragma unroll
            for (int i = 0; i < 4; ++i) {
                v2h p = { (_Float16)a0[i], (_Float16)a1[i] };
                *(v2h*)&xT[(fgrp + i) * LDS_S + lp] = p;   // ds_store_b32, transposed
            }
            if (l0 + 32 < L_)
                __builtin_prefetch(xbase + (size_t)(l0 + 32 + lp) * F_ + fgrp, 0, 0);
        }
        __syncthreads();

        // A fragment (global, L2-resident: 64KB per n reused by 32 blocks)
        v8h alo = *(const v8h*)(abase + l0);
        v8h ahi = *(const v8h*)(abase + l0 + 8);
        v16h afrag = __builtin_shufflevector(alo, ahi,
                       0,1,2,3,4,5,6,7,8,9,10,11,12,13,14,15);

        // B fragments (LDS, 16B aligned b128 loads)
        v8h b0lo = *(const v8h*)(bptr0);
        v8h b0hi = *(const v8h*)(bptr0 + 8);
        v16h bf0 = __builtin_shufflevector(b0lo, b0hi,
                       0,1,2,3,4,5,6,7,8,9,10,11,12,13,14,15);
        v8h b1lo = *(const v8h*)(bptr1);
        v8h b1hi = *(const v8h*)(bptr1 + 8);
        v16h bf1 = __builtin_shufflevector(b1lo, b1hi,
                       0,1,2,3,4,5,6,7,8,9,10,11,12,13,14,15);

        acc0 = __builtin_amdgcn_wmma_f32_16x16x32_f16(
                   false, afrag, false, bf0, (short)0, acc0, false, false);
        acc1 = __builtin_amdgcn_wmma_f32_16x16x32_f16(
                   false, afrag, false, bf1, (short)0, acc1, false, false);
    }

    // Epilogue: D layout -> lane<16: (M=m0+v, N=lane); lane>=16: (M=m0+8+v, N=lane-16)
    float* obase = out + (size_t)bn * M_ * F_;
    const int fc  = lane & 15;
    const int mhi = (lane >> 4) * 8;
#pragma unroll
    for (int v = 0; v < 8; ++v) {
        const int m  = m0 + mhi + v;
        const float r = recip[n * M_ + m];
        obase[(size_t)m * F_ + f0 + fc]      = acc0[v] * r;
        obase[(size_t)m * F_ + f0 + 16 + fc] = acc1[v] * r;
    }
}

// ---------------------------------------------------------------------------
extern "C" void kernel_launch(void* const* d_in, const int* in_sizes, int n_in,
                              void* d_out, int out_size, void* d_ws, size_t ws_size,
                              hipStream_t stream)
{
    (void)in_sizes; (void)n_in; (void)out_size; (void)ws_size;

    const float* x_aug   = (const float*)d_in[0];   // (B,N,L,F)
    const float* t       = (const float*)d_in[1];   // (L,1)
    const float* t_left  = (const float*)d_in[2];   // (N,M)
    const float* t_right = (const float*)d_in[3];   // (N,M)
    const float* kappa   = (const float*)d_in[4];   // (N,1)
    float* outp          = (float*)d_out;           // (B,N,M,F) f32

    // Workspace: alphaT (N*M*L f16 = 2 MB) then recip (N*M f32 = 8 KB)
    _Float16* alphaT = (_Float16*)d_ws;
    float*    recip  = (float*)((char*)d_ws + (size_t)N_ * M_ * L_ * sizeof(_Float16));

    wagg_alpha_kernel<<<dim3(N_ * M_), dim3(256), 0, stream>>>(
        t, t_left, t_right, kappa, alphaT, recip);

    wagg_gemm_kernel<<<dim3(B_ * N_), dim3(256), 0, stream>>>(
        x_aug, alphaT, recip, outp);
}
